// Encoder_23381801959708
// MI455X (gfx1250) — matrix operand order repaired and checked by reading, not compiled, not run on hardware
//
#include <hip/hip_runtime.h>
#include <hip/hip_bf16.h>

#define BDIM  32
#define TDIM  128
#define DDIM  512
#define FFD   2048
#define NEXP  8
#define NLAY  3
#define NFF   16
#define NFEAT 200
#define MTOK  (BDIM*TDIM)   // 4096

typedef __attribute__((ext_vector_type(16))) __bf16 v16bf;
typedef __attribute__((ext_vector_type(8)))  __bf16 v8bf;
typedef __attribute__((ext_vector_type(4)))  __bf16 v4bf;
typedef __attribute__((ext_vector_type(2)))  __bf16 v2bf;
typedef __attribute__((ext_vector_type(8)))  float  v8f;

// Two f32 -> packed bf16 (RNE), kept as a <2 x bf16> vector (no scalarization).
__device__ __forceinline__ v2bf cvt2bf(float a, float b) {
#if __has_builtin(__builtin_amdgcn_cvt_pk_bf16_f32)
  return __builtin_amdgcn_cvt_pk_bf16_f32(a, b);
#else
  v2bf r; r[0] = (__bf16)a; r[1] = (__bf16)b; return r;
#endif
}

// ---------------------------------------------------------------------------
// bf16-WMMA GEMM: C[M,N](f32) = A[M,K](f32) @ B[K,N](f32) + epilogue
// Block tile 128x128, BK=32, 256 threads (8 waves), wave tile 64x32.
// GATHER: 0 = direct A rows, 1 = A rows gathered via idx[]
// EPI bit0 = ReLU; bit1 = scatter: scatC[idx[R]] += scale[R]*v (atomic)
// KGUARD: 1 = K not multiple of 32 (element guards); 0 = fast path
// ---------------------------------------------------------------------------
template<int GATHER, int EPI, int KGUARD>
__global__ __launch_bounds__(256)
void gemm_bf16_wmma(
    const float* __restrict__ A,  long long sA,
    const float* __restrict__ Bw, long long sB,
    float* __restrict__ C,        long long sC,
    const float* __restrict__ bias,   long long sBias,
    const float* __restrict__ addvec, long long sAdd,
    const int*   __restrict__ idx,    long long sIdx,
    const int*   __restrict__ cnt,
    const float* __restrict__ scale,  long long sScale,
    float* __restrict__ scatC,
    int M, int N, int K, int lda, int ldb, int ldc)
{
  const int z  = blockIdx.z;
  const int m0 = blockIdx.y * 128;
  const int n0 = blockIdx.x * 128;
  const int mcnt = cnt ? cnt[z] : M;
  if (m0 >= mcnt) return;

  A  += z * sA;  Bw += z * sB;
  float* Cp = C ? C + z * sC : nullptr;
  const float* biasp = bias   ? bias   + z * sBias  : nullptr;
  const float* addp  = addvec ? addvec + z * sAdd   : nullptr;
  const int*   idxp  = idx    ? idx    + z * sIdx   : nullptr;
  const float* sclp  = scale  ? scale  + z * sScale : nullptr;

  __shared__ __align__(16) __bf16 As[128 * 40];
  __shared__ __align__(16) __bf16 Bs[128 * 40];

  const int t    = threadIdx.x;
  const int lane = t & 31;
  const int wave = t >> 5;
  const int half = lane >> 4;
  const int l15  = lane & 15;
  const int wm   = (wave >> 2) * 64;   // 0 / 64
  const int wn   = (wave & 3) * 32;    // 0,32,64,96

  v8f acc[4][2];
  #pragma unroll
  for (int mi = 0; mi < 4; ++mi)
    #pragma unroll
    for (int ni = 0; ni < 2; ++ni)
      #pragma unroll
      for (int q = 0; q < 8; ++q) acc[mi][ni][q] = 0.f;

  // ---- per-thread staging geometry (hoisted) ----
  const int a_r0 = t >> 3;            // 0..31
  const int a_c0 = (t & 7) * 4;       // 0..28
  const int b_n  = t & 127;           // 0..127
  const int b_kh = (t >> 7) * 16;     // 0 / 16

  const float* aRow[4];
  #pragma unroll
  for (int q = 0; q < 4; ++q) {
    int gm = m0 + a_r0 + q * 32;
    int grow;
    if (GATHER == 1) grow = (gm < mcnt) ? idxp[gm] : 0;   // clamp (masked later)
    else             grow = gm;                            // M rows always resident
    aRow[q] = A + (long long)grow * lda + a_c0;
  }
  const float* bCol = Bw + n0 + b_n + (long long)b_kh * ldb;

  float4 aReg[4];
  float  bReg[16];

  // ---- prefetch helpers ----
  auto loadA = [&](int kt) {
    #pragma unroll
    for (int q = 0; q < 4; ++q) {
      if (KGUARD) {
        float4 f;
        f.x = (kt + a_c0 + 0 < K) ? aRow[q][kt + 0] : 0.f;
        f.y = (kt + a_c0 + 1 < K) ? aRow[q][kt + 1] : 0.f;
        f.z = (kt + a_c0 + 2 < K) ? aRow[q][kt + 2] : 0.f;
        f.w = (kt + a_c0 + 3 < K) ? aRow[q][kt + 3] : 0.f;
        aReg[q] = f;
      } else {
        aReg[q] = *(const float4*)(aRow[q] + kt);
      }
    }
  };
  auto loadB = [&](int kt) {
    #pragma unroll
    for (int kk = 0; kk < 16; ++kk) {
      if (KGUARD)
        bReg[kk] = (kt + b_kh + kk < K) ? bCol[(long long)kt * ldb + (long long)kk * ldb] : 0.f;
      else
        bReg[kk] = bCol[(long long)(kt + kk) * ldb];
    }
  };
  auto storeTiles = [&]() {
    #pragma unroll
    for (int q = 0; q < 4; ++q) {
      v2bf lo = cvt2bf(aReg[q].x, aReg[q].y);
      v2bf hi = cvt2bf(aReg[q].z, aReg[q].w);
      v4bf v  = __builtin_shufflevector(lo, hi, 0, 1, 2, 3);
      *(v4bf*)&As[(a_r0 + q * 32) * 40 + a_c0] = v;       // 8B ds_store
    }
    #pragma unroll
    for (int kk = 0; kk < 4; ++kk) {
      v2bf p0 = cvt2bf(bReg[4 * kk + 0], bReg[4 * kk + 1]);
      v2bf p1 = cvt2bf(bReg[4 * kk + 2], bReg[4 * kk + 3]);
      v4bf v  = __builtin_shufflevector(p0, p1, 0, 1, 2, 3);
      *(v4bf*)&Bs[b_n * 40 + b_kh + 4 * kk] = v;          // 8B ds_store
    }
  };

  loadA(0); loadB(0);

  for (int kt = 0; kt < K; kt += 32) {
    storeTiles();
    __syncthreads();
    if (kt + 32 < K) { loadA(kt + 32); loadB(kt + 32); }  // prefetch next tile

    union Frag { v16bf v; v8bf h[2]; };
    Frag fa[4], fb[2];
    #pragma unroll
    for (int mi = 0; mi < 4; ++mi) {
      const __bf16* p = &As[(wm + mi * 16 + l15) * 40 + half * 8];
      fa[mi].h[0] = *(const v8bf*)p;
      fa[mi].h[1] = *(const v8bf*)(p + 16);
    }
    #pragma unroll
    for (int ni = 0; ni < 2; ++ni) {
      const __bf16* p = &Bs[(wn + ni * 16 + l15) * 40 + half * 8];
      fb[ni].h[0] = *(const v8bf*)p;
      fb[ni].h[1] = *(const v8bf*)(p + 16);
    }
    #pragma unroll
    for (int mi = 0; mi < 4; ++mi)
      #pragma unroll
      for (int ni = 0; ni < 2; ++ni)
        acc[mi][ni] = __builtin_amdgcn_wmma_f32_16x16x32_bf16(
            false, fa[mi].v, false, fb[ni].v, (short)0, acc[mi][ni], false, false);
    __syncthreads();
  }

  // ---- epilogue ----
  #pragma unroll
  for (int ni = 0; ni < 2; ++ni) {
    int colg = n0 + wn + ni * 16 + l15;
    float addc = 0.f;
    if (biasp) addc += biasp[colg];
    if (addp)  addc += addp[colg];
    #pragma unroll
    for (int mi = 0; mi < 4; ++mi) {
      #pragma unroll
      for (int j = 0; j < 8; ++j) {
        int R = m0 + wm + mi * 16 + half * 8 + j;
        float v = acc[mi][ni][j] + addc;
        if (EPI & 1) v = fmaxf(v, 0.f);
        if (R < mcnt) {
          if (EPI & 2) {
            int tok = idxp[R];
            float s = sclp ? sclp[R] : 1.f;
            atomicAdd(&scatC[(long long)tok * ldc + colg], s * v);
          } else {
            Cp[(long long)R * ldc + colg] = v;
          }
        }
      }
    }
  }
}

// ---------------------------------------------------------------------------
// pb[b,i,j] = sum_f a[f] * cos((frac[b,j]-frac[b,i])*10 * W[f] + b[f])
// ---------------------------------------------------------------------------
__global__ void fourier_pb_kernel(const float* __restrict__ frac,
                                  const float* __restrict__ pbW,
                                  const float* __restrict__ pbb,
                                  const float* __restrict__ pba,
                                  float* __restrict__ pb)
{
  int id = blockIdx.x * blockDim.x + threadIdx.x;
  if (id >= BDIM * TDIM * TDIM) return;
  int j = id & (TDIM - 1);
  int i = (id >> 7) & (TDIM - 1);
  int b = id >> 14;
  float d = (frac[b * TDIM + j] - frac[b * TDIM + i]) * 10.0f;
  float s = 0.f;
  #pragma unroll
  for (int f = 0; f < NFF; ++f)
    s += pba[f] * __cosf(d * pbW[f] + pbb[f]);
  pb[id] = s;
}

// kv[b,d] = sum_t K[b,t,d]*V[b,t,d]
__global__ void kv_reduce_kernel(const float* __restrict__ Kb,
                                 const float* __restrict__ Vb,
                                 float* __restrict__ kv)
{
  int id = blockIdx.x * blockDim.x + threadIdx.x;
  if (id >= BDIM * DDIM) return;
  int b = id / DDIM, d = id % DDIM;
  const float* kp = Kb + (long long)b * TDIM * DDIM + d;
  const float* vp = Vb + (long long)b * TDIM * DDIM + d;
  float s = 0.f;
  for (int tt = 0; tt < TDIM; ++tt) s += kp[tt * DDIM] * vp[tt * DDIM];
  kv[id] = s;
}

// logits[tok,e] = x[tok,:] . gW[:,e] + gb[e]
__global__ void gate_logits_kernel(const float* __restrict__ x,
                                   const float* __restrict__ gW,
                                   const float* __restrict__ gb,
                                   float* __restrict__ logits)
{
  int id = blockIdx.x * blockDim.x + threadIdx.x;
  if (id >= MTOK * NEXP) return;
  int e = id & (NEXP - 1), tok = id >> 3;
  const float* xp = x + (long long)tok * DDIM;
  float s = gb[e];
  for (int d = 0; d < DDIM; ++d) s += xp[d] * gW[d * NEXP + e];
  logits[id] = s;
}

// softmax + top-2 + scatter into per-expert token lists
__global__ void top2_scatter_kernel(const float* __restrict__ logits,
                                    int* __restrict__ cnt,
                                    int* __restrict__ eidx,
                                    float* __restrict__ ewgt)
{
  int tok = blockIdx.x * blockDim.x + threadIdx.x;
  if (tok >= MTOK) return;
  float l[NEXP], p[NEXP];
  float m = -1e30f;
  #pragma unroll
  for (int e = 0; e < NEXP; ++e) { l[e] = logits[tok * NEXP + e]; m = fmaxf(m, l[e]); }
  float sum = 0.f;
  #pragma unroll
  for (int e = 0; e < NEXP; ++e) { p[e] = __expf(l[e] - m); sum += p[e]; }
  float inv = 1.f / sum;
  #pragma unroll
  for (int e = 0; e < NEXP; ++e) p[e] *= inv;
  int e1 = 0;
  #pragma unroll
  for (int e = 1; e < NEXP; ++e) if (p[e] > p[e1]) e1 = e;
  int e2 = (e1 == 0) ? 1 : 0;
  #pragma unroll
  for (int e = 0; e < NEXP; ++e) if (e != e1 && p[e] > p[e2]) e2 = e;
  int pos1 = atomicAdd(&cnt[e1], 1);
  eidx[e1 * MTOK + pos1] = tok; ewgt[e1 * MTOK + pos1] = p[e1];
  int pos2 = atomicAdd(&cnt[e2], 1);
  eidx[e2 * MTOK + pos2] = tok; ewgt[e2 * MTOK + pos2] = p[e2];
}

__global__ void zero_u32_kernel(unsigned* __restrict__ p, long long n)
{
  long long id = (long long)blockIdx.x * blockDim.x + threadIdx.x;
  if (id < n) p[id] = 0u;
}

// out[row,:] = LN(x[row,:] + y[row,:]; g, b), one block per row, D=512
__global__ __launch_bounds__(256)
void ln_add_kernel(const float* __restrict__ x, const float* __restrict__ y,
                   const float* __restrict__ g, const float* __restrict__ b,
                   float* __restrict__ out)
{
  __shared__ float red[256];
  int row = blockIdx.x;
  int t = threadIdx.x;
  const float* xp = x + (long long)row * DDIM;
  const float* yp = y + (long long)row * DDIM;
  float v0 = xp[t]       + yp[t];
  float v1 = xp[t + 256] + yp[t + 256];
  red[t] = v0 + v1;
  __syncthreads();
  for (int s = 128; s > 0; s >>= 1) { if (t < s) red[t] += red[t + s]; __syncthreads(); }
  float mean = red[0] * (1.f / DDIM);
  __syncthreads();
  float d0 = v0 - mean, d1 = v1 - mean;
  red[t] = d0 * d0 + d1 * d1;
  __syncthreads();
  for (int s = 128; s > 0; s >>= 1) { if (t < s) red[t] += red[t + s]; __syncthreads(); }
  float var = red[0] * (1.f / DDIM);
  float rs = rsqrtf(var + 1e-5f);
  float* op = out + (long long)row * DDIM;
  op[t]       = d0 * rs * g[t]       + b[t];
  op[t + 256] = d1 * rs * g[t + 256] + b[t + 256];
}

// ---------------------------------------------------------------------------
extern "C" void kernel_launch(void* const* d_in, const int* in_sizes, int n_in,
                              void* d_out, int out_size, void* d_ws, size_t ws_size,
                              hipStream_t stream)
{
  (void)in_sizes; (void)n_in; (void)out_size; (void)ws_size;
  const int*   src  = (const int*)d_in[0];
  const float* frac = (const float*)d_in[1];
  const float* etab = (const float*)d_in[2];
  const float* Wm2v = (const float*)d_in[3];
  const float* bm2v = (const float*)d_in[4];
  const float* pbW  = (const float*)d_in[5];
  const float* pbb  = (const float*)d_in[6];
  const float* pba  = (const float*)d_in[7];
  const float* Wk   = (const float*)d_in[8];
  const float* bk   = (const float*)d_in[9];
  const float* Wv   = (const float*)d_in[10];
  const float* bv   = (const float*)d_in[11];
  const float* Wo   = (const float*)d_in[12];
  const float* bo   = (const float*)d_in[13];
  const float* ln1g = (const float*)d_in[14];
  const float* ln1b = (const float*)d_in[15];
  const float* ln2g = (const float*)d_in[16];
  const float* ln2b = (const float*)d_in[17];
  const float* gW   = (const float*)d_in[18];
  const float* gb   = (const float*)d_in[19];
  const float* eW1  = (const float*)d_in[20];
  const float* eb1  = (const float*)d_in[21];
  const float* eW2  = (const float*)d_in[22];
  const float* eb2  = (const float*)d_in[23];

  char* w = (char*)d_ws;
  auto alloc = [&](size_t bytes) {
    char* p = w; w += (bytes + 255) & ~(size_t)255; return p;
  };
  float* x    = (float*)alloc((size_t)MTOK * DDIM * 4);
  float* Kb   = (float*)alloc((size_t)MTOK * DDIM * 4);
  float* Vb   = (float*)alloc((size_t)MTOK * DDIM * 4);
  float* pb   = (float*)alloc((size_t)BDIM * TDIM * TDIM * 4);
  float* kvb  = (float*)alloc((size_t)BDIM * DDIM * 4);
  float* wv   = (float*)alloc((size_t)MTOK * DDIM * 4);
  float* t2   = (float*)alloc((size_t)MTOK * DDIM * 4);
  float* lgts = (float*)alloc((size_t)MTOK * NEXP * 4);
  int*   cnt  = (int*)  alloc((size_t)NEXP * 4);
  int*   eidx = (int*)  alloc((size_t)NEXP * MTOK * 4);
  float* ewgt = (float*)alloc((size_t)NEXP * MTOK * 4);
  float* ff   = (float*)alloc((size_t)MTOK * DDIM * 4);
  float* hbuf = (float*)alloc((size_t)MTOK * FFD * 4);

  // Fourier positional bias (used by all layers)
  fourier_pb_kernel<<<(BDIM * TDIM * TDIM + 255) / 256, 256, 0, stream>>>(
      frac, pbW, pbb, pba, pb);

  // Embedding: x = etab[src] @ Wm2v + bm2v  (gathered-A GEMM, K=200 zero-padded)
  {
    dim3 g(DDIM / 128, MTOK / 128, 1);
    gemm_bf16_wmma<1, 0, 1><<<g, 256, 0, stream>>>(
        etab, 0, Wm2v, 0, x, 0, bm2v, 0, nullptr, 0,
        src, 0, nullptr, nullptr, 0, nullptr,
        MTOK, DDIM, NFEAT, NFEAT, DDIM, DDIM);
  }

  for (int i = 0; i < NLAY; ++i) {
    const float* Wki = Wk + (size_t)i * DDIM * DDIM;
    const float* bki = bk + (size_t)i * DDIM;
    const float* Wvi = Wv + (size_t)i * DDIM * DDIM;
    const float* bvi = bv + (size_t)i * DDIM;
    const float* Woi = Wo + (size_t)i * DDIM * DDIM;
    const float* boi = bo + (size_t)i * DDIM;

    dim3 g4(DDIM / 128, MTOK / 128, 1);
    // K, V projections
    gemm_bf16_wmma<0, 0, 0><<<g4, 256, 0, stream>>>(
        x, 0, Wki, 0, Kb, 0, bki, 0, nullptr, 0,
        nullptr, 0, nullptr, nullptr, 0, nullptr,
        MTOK, DDIM, DDIM, DDIM, DDIM, DDIM);
    gemm_bf16_wmma<0, 0, 0><<<g4, 256, 0, stream>>>(
        x, 0, Wvi, 0, Vb, 0, bvi, 0, nullptr, 0,
        nullptr, 0, nullptr, nullptr, 0, nullptr,
        MTOK, DDIM, DDIM, DDIM, DDIM, DDIM);
    // kv[b,d] = sum_t K*V
    kv_reduce_kernel<<<(BDIM * DDIM + 255) / 256, 256, 0, stream>>>(Kb, Vb, kvb);
    // wv[b] = pb[b] @ V[b] + kv[b,:]  (batched GEMM, z = 32)
    {
      dim3 gp(DDIM / 128, 1, BDIM);
      gemm_bf16_wmma<0, 0, 0><<<gp, 256, 0, stream>>>(
          pb, (long long)TDIM * TDIM, Vb, (long long)TDIM * DDIM,
          wv, (long long)TDIM * DDIM, nullptr, 0, kvb, DDIM,
          nullptr, 0, nullptr, nullptr, 0, nullptr,
          TDIM, DDIM, TDIM, TDIM, DDIM, DDIM);
    }
    // t2 = wv @ Wo + bo
    gemm_bf16_wmma<0, 0, 0><<<g4, 256, 0, stream>>>(
        wv, 0, Woi, 0, t2, 0, boi, 0, nullptr, 0,
        nullptr, 0, nullptr, nullptr, 0, nullptr,
        MTOK, DDIM, DDIM, DDIM, DDIM, DDIM);
    // x = LN1(x + t2)
    ln_add_kernel<<<MTOK, 256, 0, stream>>>(
        x, t2, ln1g + (size_t)i * DDIM, ln1b + (size_t)i * DDIM, x);

    // MoE gate
    gate_logits_kernel<<<(MTOK * NEXP + 255) / 256, 256, 0, stream>>>(
        x, gW + (size_t)i * DDIM * NEXP, gb + (size_t)i * NEXP, lgts);
    zero_u32_kernel<<<1, 256, 0, stream>>>((unsigned*)cnt, NEXP);
    zero_u32_kernel<<<(MTOK * DDIM + 255) / 256, 256, 0, stream>>>(
        (unsigned*)ff, (long long)MTOK * DDIM);
    top2_scatter_kernel<<<(MTOK + 255) / 256, 256, 0, stream>>>(lgts, cnt, eidx, ewgt);

    // Experts (serialized; stream order protects shared hbuf)
    for (int e = 0; e < NEXP; ++e) {
      const float* W1 = eW1 + ((size_t)i * NEXP + e) * DDIM * FFD;
      const float* b1 = eb1 + ((size_t)i * NEXP + e) * FFD;
      const float* W2 = eW2 + ((size_t)i * NEXP + e) * FFD * DDIM;
      const float* b2 = eb2 + ((size_t)i * NEXP + e) * DDIM;
      // h = relu(gather(x, idx_e) @ W1 + b1), compact rows
      dim3 g1(FFD / 128, MTOK / 128, 1);
      gemm_bf16_wmma<1, 1, 0><<<g1, 256, 0, stream>>>(
          x, 0, W1, 0, hbuf, 0, b1, 0, nullptr, 0,
          eidx + (size_t)e * MTOK, 0, cnt + e, nullptr, 0, nullptr,
          MTOK, FFD, DDIM, DDIM, FFD, FFD);
      // ff[idx_e[m]] += wgt[m] * (h @ W2 + b2)   (atomic scatter)
      dim3 g2(DDIM / 128, MTOK / 128, 1);
      gemm_bf16_wmma<0, 2, 0><<<g2, 256, 0, stream>>>(
          hbuf, 0, W2, 0, nullptr, 0, b2, 0, nullptr, 0,
          eidx + (size_t)e * MTOK, 0, cnt + e, ewgt + (size_t)e * MTOK, 0, ff,
          MTOK, DDIM, FFD, FFD, DDIM, DDIM);
    }
    // x = LN2(x + ff); final layer writes d_out
    float* dst = (i == NLAY - 1) ? (float*)d_out : x;
    ln_add_kernel<<<MTOK, 256, 0, stream>>>(
        x, ff, ln2g + (size_t)i * DDIM, ln2b + (size_t)i * DDIM, dst);
  }
}
